// Model_36979668418647
// MI455X (gfx1250) — matrix operand order repaired and checked by reading, not compiled, not run on hardware
//
#include <hip/hip_runtime.h>

// ---------------------------------------------------------------------------
// Types for gfx1250 WMMA (wave32)
// ---------------------------------------------------------------------------
typedef __attribute__((ext_vector_type(16))) __bf16        v16bf;
typedef __attribute__((ext_vector_type(8)))  __bf16        v8bf;
typedef __attribute__((ext_vector_type(8)))  float         v8f;
typedef __attribute__((ext_vector_type(8)))  unsigned short us8;

#define BATCH   32768
#define BM      128
#define BN      128     // sync kernel block-N
#define BNA     256     // async kernel block-N
#define BK      32
#define LDT     40      // LDS row stride (ushort): 32 + 8 pad, keeps 16B alignment
#define NSPLIT  64
#define STATS_STRIDE 1024

__device__ __forceinline__ unsigned short f2bf_rne(float f) {
    unsigned int u = __float_as_uint(f);
    u += 0x7FFFu + ((u >> 16) & 1u);
    return (unsigned short)(u >> 16);
}
__device__ __forceinline__ float bf2f(unsigned short h) {
    return __uint_as_float(((unsigned int)h) << 16);
}

// ---------------------------------------------------------------------------
// 16B global -> LDS copy. On gfx1250 use the async-tensor path
// (GLOBAL_LOAD_ASYNC_TO_LDS_B128, tracked by ASYNCcnt). LDS byte address is
// the low 32 bits of the generic pointer.
// ---------------------------------------------------------------------------
__device__ __forceinline__ void cp_g2l_b128(const unsigned short* g, unsigned short* l) {
#if defined(__gfx1250__)
    unsigned int ldsoff = (unsigned int)(unsigned long long)(void*)l;
    asm volatile("global_load_async_to_lds_b128 %0, %1, off"
                 :
                 : "v"(ldsoff), "v"(g)
                 : "memory");
#else
    *(us8*)l = *(const us8*)g;
#endif
}
__device__ __forceinline__ void wait_async_lds() {
#if defined(__gfx1250__)
    asm volatile("s_wait_asynccnt 0" ::: "memory");
#endif
}

// ---------------------------------------------------------------------------
// Async double-buffered bf16 WMMA GEMM:  C[M,N] = A[M,K] * W[N,K]^T + bias
//   A : bf16 row-major, lda == K (multiple of 32)
//   W : bf16 row-major [N, K] pre-padded
//   C : fp32 or bf16
// Block 256 threads = 8 waves; block tile 128x256; wave tile 64x64 (4x4 WMMA).
// ---------------------------------------------------------------------------
__global__ __launch_bounds__(256)
void gemm_bf16_wmma_async(const unsigned short* __restrict__ A, int lda,
                          const unsigned short* __restrict__ Wt, int ldw,
                          const float* __restrict__ bias,
                          void* __restrict__ Cv, int cIsF32, int ldc,
                          int N, int K)
{
    __shared__ unsigned short ldsA[2][BM  * LDT];
    __shared__ unsigned short ldsB[2][BNA * LDT];

    const int tid  = threadIdx.x;
    const int lane = tid & 31;
    const int wave = tid >> 5;
    const int wm   = wave >> 2;   // 0..1 : 64-row slice
    const int wn   = wave & 3;    // 0..3 : 64-col slice
    const int l16  = lane & 15;
    const int hi   = lane >> 4;

    const long blockM = (long)blockIdx.y * BM;
    const int  blockN = blockIdx.x * BNA;

    v8f acc[4][4];
#pragma unroll
    for (int i = 0; i < 4; ++i)
#pragma unroll
        for (int j = 0; j < 4; ++j)
#pragma unroll
            for (int r = 0; r < 8; ++r) acc[i][j][r] = 0.0f;

    auto issue_tile = [&](int buf, int kb) {
        // A tile: 128 rows x 32 K = 512 16B-chunks, 2 per thread
#pragma unroll
        for (int c = 0; c < 2; ++c) {
            int  id  = tid + c * 256;
            int  row = id >> 2;
            int  c8  = (id & 3) * 8;
            long gr  = blockM + row;
            cp_g2l_b128(A + gr * (long)lda + kb + c8,
                        &ldsA[buf][row * LDT + c8]);
        }
        // B tile: 256 N-rows x 32 K = 1024 chunks, 4 per thread
#pragma unroll
        for (int c = 0; c < 4; ++c) {
            int id  = tid + c * 256;
            int row = id >> 2;
            int c8  = (id & 3) * 8;
            int gn  = blockN + row;
            if (gn < N) {
                cp_g2l_b128(Wt + (long)gn * ldw + kb + c8,
                            &ldsB[buf][row * LDT + c8]);
            } else {
                us8 z;
#pragma unroll
                for (int j = 0; j < 8; ++j) z[j] = 0;
                *(us8*)&ldsB[buf][row * LDT + c8] = z;
            }
        }
    };

    const int nk = K / BK;
    issue_tile(0, 0);

    for (int t = 0; t < nk; ++t) {
        const int cur = t & 1;
        wait_async_lds();
        __syncthreads();                 // tile `cur` complete & visible
        if (t + 1 < nk) issue_tile(cur ^ 1, (t + 1) * BK);

        // A 16x32 bf16 fragments: lane-half selects K 8-groups
        v16bf afrag[4];
#pragma unroll
        for (int mt = 0; mt < 4; ++mt) {
            int m = wm * 64 + mt * 16 + l16;
            v8bf lo = *(const v8bf*)&ldsA[cur][m * LDT + hi * 8];
            v8bf hh = *(const v8bf*)&ldsA[cur][m * LDT + 16 + hi * 8];
            afrag[mt] = __builtin_shufflevector(lo, hh,
                0,1,2,3,4,5,6,7,8,9,10,11,12,13,14,15);
        }
#pragma unroll
        for (int nt = 0; nt < 4; ++nt) {
            int n = wn * 64 + nt * 16 + l16;
            // B 32x16 bf16: lane-half selects K 8-groups (same order as A)
            v8bf lo = *(const v8bf*)&ldsB[cur][n * LDT + hi * 8];
            v8bf hh = *(const v8bf*)&ldsB[cur][n * LDT + 16 + hi * 8];
            v16bf bfrag = __builtin_shufflevector(lo, hh,
                0,1,2,3,4,5,6,7,8,9,10,11,12,13,14,15);
#pragma unroll
            for (int mt = 0; mt < 4; ++mt) {
                acc[mt][nt] = __builtin_amdgcn_wmma_f32_16x16x32_bf16(
                    false, afrag[mt], false, bfrag,
                    (short)0, acc[mt][nt], false, false);
            }
        }
        __syncthreads();                 // all reads of `cur` done before overwrite
    }

    float*          Cf = (float*)Cv;
    unsigned short* Ch = (unsigned short*)Cv;
#pragma unroll
    for (int mt = 0; mt < 4; ++mt) {
#pragma unroll
        for (int nt = 0; nt < 4; ++nt) {
            int col = blockN + wn * 64 + nt * 16 + l16;
            if (col < N) {
                float bv = bias ? bias[col] : 0.0f;
#pragma unroll
                for (int r = 0; r < 8; ++r) {
                    long row = blockM + wm * 64 + mt * 16 + r + 8 * hi;
                    float val = acc[mt][nt][r] + bv;
                    if (cIsF32) Cf[row * (long)ldc + col] = val;
                    else        Ch[row * (long)ldc + col] = f2bf_rne(val);
                }
            }
        }
    }
}

// ---------------------------------------------------------------------------
// Synchronous GEMM with on-the-fly fp32->bf16 A conversion (first layers).
// Block tile 128x128; wave tile 64x32 (4x2 WMMA).
// ---------------------------------------------------------------------------
__global__ __launch_bounds__(256)
void gemm_f32a_wmma(const float* __restrict__ Af, int lda, int KA,
                    const unsigned short* __restrict__ Wt, int ldw,
                    const float* __restrict__ bias,
                    void* __restrict__ Cv, int cIsF32, int ldc,
                    int N, int K)
{
    __shared__ unsigned short ldsA[BM * LDT];
    __shared__ unsigned short ldsB[BN * LDT];

    const int tid  = threadIdx.x;
    const int lane = tid & 31;
    const int wave = tid >> 5;
    const int wm   = wave >> 2;
    const int wn   = wave & 3;
    const int l16  = lane & 15;
    const int hi   = lane >> 4;

    const long blockM = (long)blockIdx.y * BM;
    const int  blockN = blockIdx.x * BN;

    v8f acc[4][2];
#pragma unroll
    for (int i = 0; i < 4; ++i)
#pragma unroll
        for (int j = 0; j < 2; ++j)
#pragma unroll
            for (int r = 0; r < 8; ++r) acc[i][j][r] = 0.0f;

    for (int kb = 0; kb < K; kb += BK) {
#pragma unroll
        for (int c = 0; c < 2; ++c) {
            int  id  = tid + c * 256;
            int  row = id >> 2;
            int  c8  = (id & 3) * 8;
            long gr  = blockM + row;
            int  k0  = kb + c8;
            us8  v;
            if (k0 + 8 <= KA) {
                const float* p = Af + gr * (long)lda + k0;
                float4 f0 = *(const float4*)p;
                float4 f1 = *(const float4*)(p + 4);
                v[0] = f2bf_rne(f0.x); v[1] = f2bf_rne(f0.y);
                v[2] = f2bf_rne(f0.z); v[3] = f2bf_rne(f0.w);
                v[4] = f2bf_rne(f1.x); v[5] = f2bf_rne(f1.y);
                v[6] = f2bf_rne(f1.z); v[7] = f2bf_rne(f1.w);
            } else {
#pragma unroll
                for (int j = 0; j < 8; ++j) {
                    int k = k0 + j;
                    v[j] = (k < KA) ? f2bf_rne(Af[gr * (long)lda + k])
                                    : (unsigned short)0;
                }
            }
            *(us8*)&ldsA[row * LDT + c8] = v;
        }
#pragma unroll
        for (int c = 0; c < 2; ++c) {
            int id  = tid + c * 256;
            int row = id >> 2;
            int c8  = (id & 3) * 8;
            int gn  = blockN + row;
            us8 v;
            if (gn < N) {
                v = *(const us8*)(Wt + (long)gn * ldw + kb + c8);
            } else {
#pragma unroll
                for (int j = 0; j < 8; ++j) v[j] = 0;
            }
            *(us8*)&ldsB[row * LDT + c8] = v;
        }
        __syncthreads();

        v16bf afrag[4];
#pragma unroll
        for (int mt = 0; mt < 4; ++mt) {
            int m = wm * 64 + mt * 16 + l16;
            v8bf lo = *(const v8bf*)&ldsA[m * LDT + hi * 8];
            v8bf hh = *(const v8bf*)&ldsA[m * LDT + 16 + hi * 8];
            afrag[mt] = __builtin_shufflevector(lo, hh,
                0,1,2,3,4,5,6,7,8,9,10,11,12,13,14,15);
        }
#pragma unroll
        for (int nt = 0; nt < 2; ++nt) {
            int n = wn * 32 + nt * 16 + l16;
            v8bf lo = *(const v8bf*)&ldsB[n * LDT + hi * 8];
            v8bf hh = *(const v8bf*)&ldsB[n * LDT + 16 + hi * 8];
            v16bf bfrag = __builtin_shufflevector(lo, hh,
                0,1,2,3,4,5,6,7,8,9,10,11,12,13,14,15);
#pragma unroll
            for (int mt = 0; mt < 4; ++mt) {
                acc[mt][nt] = __builtin_amdgcn_wmma_f32_16x16x32_bf16(
                    false, afrag[mt], false, bfrag,
                    (short)0, acc[mt][nt], false, false);
            }
        }
        __syncthreads();
    }

    float*          Cf = (float*)Cv;
    unsigned short* Ch = (unsigned short*)Cv;
#pragma unroll
    for (int mt = 0; mt < 4; ++mt) {
#pragma unroll
        for (int nt = 0; nt < 2; ++nt) {
            int col = blockN + wn * 32 + nt * 16 + l16;
            if (col < N) {
                float bv = bias ? bias[col] : 0.0f;
#pragma unroll
                for (int r = 0; r < 8; ++r) {
                    long row = blockM + wm * 64 + mt * 16 + r + 8 * hi;
                    float val = acc[mt][nt][r] + bv;
                    if (cIsF32) Cf[row * (long)ldc + col] = val;
                    else        Ch[row * (long)ldc + col] = f2bf_rne(val);
                }
            }
        }
    }
}

// ---------------------------------------------------------------------------
// fp32 weight [N,Kin] -> bf16 [N,Kpad] (zero pad K)
// ---------------------------------------------------------------------------
__global__ void convert_weight(const float* __restrict__ W,
                               unsigned short* __restrict__ out,
                               int N, int Kin, int Kpad)
{
    long i = (long)blockIdx.x * blockDim.x + threadIdx.x;
    long total = (long)N * Kpad;
    if (i >= total) return;
    int  k = (int)(i % Kpad);
    long n = i / Kpad;
    out[i] = (k < Kin) ? f2bf_rne(W[n * (long)Kin + k]) : (unsigned short)0;
}

// ---------------------------------------------------------------------------
// Deterministic column stats, pass 1: fixed 64 contiguous row-splits
// ---------------------------------------------------------------------------
__global__ void stats_partial(const void* __restrict__ Yv, int yIsF32, int ldy,
                              int Brows, int N,
                              float* __restrict__ psum, float* __restrict__ psq)
{
    int col   = blockIdx.x * 256 + threadIdx.x;
    int split = blockIdx.y;
    if (col >= N) return;
    const float*          Yf = (const float*)Yv;
    const unsigned short* Yh = (const unsigned short*)Yv;
    long chunk = Brows / NSPLIT;
    long r0 = (long)split * chunk;
    float s = 0.f, q = 0.f;
    for (long r = r0; r < r0 + chunk; ++r) {
        float v = yIsF32 ? Yf[r * ldy + col] : bf2f(Yh[r * ldy + col]);
        s += v; q += v * v;
    }
    psum[(long)split * STATS_STRIDE + col] = s;
    psq [(long)split * STATS_STRIDE + col] = q;
}

// pass 2: ordered reduce + BN scale/shift
__global__ void stats_finalize(const float* __restrict__ psum,
                               const float* __restrict__ psq,
                               const float* __restrict__ g,
                               const float* __restrict__ beta,
                               int Brows, int N, float eps,
                               float* __restrict__ scale,
                               float* __restrict__ shift)
{
    int col = blockIdx.x * 256 + threadIdx.x;
    if (col >= N) return;
    float s = 0.f, q = 0.f;
    for (int i = 0; i < NSPLIT; ++i) {
        s += psum[(long)i * STATS_STRIDE + col];
        q += psq [(long)i * STATS_STRIDE + col];
    }
    float mean = s / (float)Brows;
    float var  = q / (float)Brows - mean * mean;
    float sc   = g[col] * rsqrtf(var + eps);
    scale[col] = sc;
    shift[col] = beta[col] - mean * sc;
}

// BN + ReLU, bf16 -> bf16
__global__ void bn_relu_bf16(const unsigned short* __restrict__ Y, int N,
                             const float* __restrict__ scale,
                             const float* __restrict__ shift,
                             unsigned short* __restrict__ out, long total)
{
    long i = (long)blockIdx.x * blockDim.x + threadIdx.x;
    if (i >= total) return;
    int col = (int)(i % N);
    float v = bf2f(Y[i]) * scale[col] + shift[col];
    out[i] = f2bf_rne(v > 0.f ? v : 0.f);
}

// loc = [lat,lon] @ Wloc^T + bloc  (K=2, N=10)
__global__ void loc_fc(const float* __restrict__ lat, const float* __restrict__ lon,
                       const float* __restrict__ Wloc, const float* __restrict__ bloc,
                       float* __restrict__ y, int Brows)
{
    int i = blockIdx.x * blockDim.x + threadIdx.x;
    if (i >= Brows) return;
    float a = lat[i], b = lon[i];
#pragma unroll
    for (int j = 0; j < 10; ++j)
        y[(long)i * 10 + j] = a * Wloc[j * 2] + b * Wloc[j * 2 + 1] + bloc[j];
}

// abuf[B,320] bf16 = [ BN/ReLU(locy) | tag | zero-pad ]
__global__ void build_a(const float* __restrict__ locy,
                        const float* __restrict__ scale,
                        const float* __restrict__ shift,
                        const float* __restrict__ tag,
                        unsigned short* __restrict__ abuf, int Brows)
{
    long i = (long)blockIdx.x * blockDim.x + threadIdx.x;
    long total = (long)Brows * 320;
    if (i >= total) return;
    int  col = (int)(i % 320);
    long row = i / 320;
    unsigned short o = 0;
    if (col < 10) {
        float v = locy[row * 10 + col] * scale[col] + shift[col];
        o = f2bf_rne(v > 0.f ? v : 0.f);
    } else if (col < 310) {
        o = f2bf_rne(tag[row * 300 + (col - 10)]);
    }
    abuf[i] = o;
}

__global__ void init_count(int* cnt)
{
    if (blockIdx.x == 0 && threadIdx.x == 0) *cnt = 0;
}

// one wave per row: pairwise distances + margin count (int atomics: deterministic)
__global__ __launch_bounds__(256)
void tri_count(const float* __restrict__ out, int Brows, int* __restrict__ cnt)
{
    int  wave = threadIdx.x >> 5;
    int  lane = threadIdx.x & 31;
    long row  = (long)blockIdx.x * 8 + wave;
    if (row >= Brows) return;
    const float* p = out;
    const float* n = out + (long)Brows * 300;
    const float* a = out + 2L * Brows * 300;
    float dp = 0.f, dn = 0.f;
    for (int c = lane; c < 300; c += 32) {
        float av = a[row * 300 + c];
        float x = fabsf(av - p[row * 300 + c] + 1e-6f);
        float y = fabsf(av - n[row * 300 + c] + 1e-6f);
        dp += x * x;
        dn += y * y;
    }
#pragma unroll
    for (int off = 16; off > 0; off >>= 1) {
        dp += __shfl_xor(dp, off, 32);
        dn += __shfl_xor(dn, off, 32);
    }
    if (lane == 0) {
        if (sqrtf(dn) - sqrtf(dp) > 0.1f) atomicAdd(cnt, 1);
    }
}

// ---------------------------------------------------------------------------
// Host-side orchestration
// ---------------------------------------------------------------------------
extern "C" void kernel_launch(void* const* d_in, const int* in_sizes, int n_in,
                              void* d_out, int out_size, void* d_ws, size_t ws_size,
                              hipStream_t stream)
{
    const int B = BATCH;
    const float EPS_BN = 1e-3f;

    const float* img_p = (const float*)d_in[0];
    const float* img_n = (const float*)d_in[1];
    const float* tag   = (const float*)d_in[2];
    const float* lat   = (const float*)d_in[3];
    const float* lon   = (const float*)d_in[4];
    const float* Wi1 = (const float*)d_in[5];  const float* bi1  = (const float*)d_in[6];
    const float* gi1 = (const float*)d_in[7];  const float* bti1 = (const float*)d_in[8];
    const float* Wi2 = (const float*)d_in[9];  const float* bi2  = (const float*)d_in[10];
    const float* Wi3 = (const float*)d_in[11]; const float* bi3  = (const float*)d_in[12];
    const float* Wloc = (const float*)d_in[13]; const float* bloc = (const float*)d_in[14];
    const float* gloc = (const float*)d_in[15]; const float* btloc= (const float*)d_in[16];
    const float* Wm1 = (const float*)d_in[17]; const float* bm1  = (const float*)d_in[18];
    const float* gm1 = (const float*)d_in[19]; const float* btm1 = (const float*)d_in[20];
    const float* Wm2 = (const float*)d_in[21]; const float* bm2  = (const float*)d_in[22];
    const float* gm2 = (const float*)d_in[23]; const float* btm2 = (const float*)d_in[24];
    const float* Wm3 = (const float*)d_in[25]; const float* bm3  = (const float*)d_in[26];
    const float* gm3 = (const float*)d_in[27]; const float* btm3 = (const float*)d_in[28];
    const float* Wm4 = (const float*)d_in[29]; const float* bm4  = (const float*)d_in[30];

    // ---- workspace carve-up ----
    char* base = (char*)d_ws;
    size_t off = 0;
    auto carve = [&](size_t bytes) -> char* {
        char* p = base + off;
        off += (bytes + 255) & ~(size_t)255;
        return p;
    };
    unsigned short* wWi1 = (unsigned short*)carve((size_t)1024 * 320 * 2);
    unsigned short* wWi2 = (unsigned short*)carve((size_t)1024 * 1024 * 2);
    unsigned short* wWi3 = (unsigned short*)carve((size_t)300 * 1024 * 2);
    unsigned short* wWm1 = (unsigned short*)carve((size_t)1024 * 320 * 2);
    unsigned short* wWm2 = (unsigned short*)carve((size_t)1024 * 1024 * 2);
    unsigned short* wWm3 = (unsigned short*)carve((size_t)1024 * 1024 * 2);
    unsigned short* wWm4 = (unsigned short*)carve((size_t)300 * 1024 * 2);
    unsigned short* act0 = (unsigned short*)carve((size_t)B * 1024 * 2);
    unsigned short* act1 = (unsigned short*)carve((size_t)B * 1024 * 2);
    unsigned short* abuf = (unsigned short*)carve((size_t)B * 320 * 2);
    float* locy  = (float*)carve((size_t)B * 10 * 4);
    float* psum  = (float*)carve((size_t)NSPLIT * STATS_STRIDE * 4);
    float* psq   = (float*)carve((size_t)NSPLIT * STATS_STRIDE * 4);
    float* scale = (float*)carve((size_t)1024 * 4);
    float* shift = (float*)carve((size_t)1024 * 4);
    (void)ws_size; (void)n_in; (void)in_sizes; (void)out_size;

    float* outP = (float*)d_out;
    float* outN = outP + (long)B * 300;
    float* outA = outP + 2L * B * 300;
    int*   cnt  = (int*)(outP + 3L * B * 300);

    auto cvtw = [&](const float* W, unsigned short* dst, int N, int Kin, int Kpad) {
        long total = (long)N * Kpad;
        hipLaunchKernelGGL(convert_weight, dim3((total + 255) / 256), dim3(256),
                           0, stream, W, dst, N, Kin, Kpad);
    };
    cvtw(Wi1, wWi1, 1024, 300, 320);
    cvtw(Wi2, wWi2, 1024, 1024, 1024);
    cvtw(Wi3, wWi3, 300, 1024, 1024);
    cvtw(Wm1, wWm1, 1024, 310, 320);
    cvtw(Wm2, wWm2, 1024, 1024, 1024);
    cvtw(Wm3, wWm3, 1024, 1024, 1024);
    cvtw(Wm4, wWm4, 300, 1024, 1024);

    // first-layer GEMMs: fp32 A converted on the fly (sync kernel)
    auto gemm_f32a = [&](const float* A, int lda, int KA,
                         const unsigned short* W, int ldw, const float* bias,
                         void* C, int cF32, int ldc, int N, int K) {
        dim3 grid((N + BN - 1) / BN, B / BM);
        hipLaunchKernelGGL(gemm_f32a_wmma, grid, dim3(256), 0, stream,
                           A, lda, KA, W, ldw, bias, C, cF32, ldc, N, K);
    };
    // bf16-A GEMMs: async double-buffered kernel
    auto gemm_async = [&](const unsigned short* A, int lda,
                          const unsigned short* W, int ldw, const float* bias,
                          void* C, int cF32, int ldc, int N, int K) {
        dim3 grid((N + BNA - 1) / BNA, B / BM);
        hipLaunchKernelGGL(gemm_bf16_wmma_async, grid, dim3(256), 0, stream,
                           A, lda, W, ldw, bias, C, cF32, ldc, N, K);
    };
    auto bnstats = [&](const void* Y, int yF32, int ldy, int N,
                       const float* g, const float* beta) {
        hipLaunchKernelGGL(stats_partial, dim3((N + 255) / 256, NSPLIT), dim3(256),
                           0, stream, Y, yF32, ldy, B, N, psum, psq);
        hipLaunchKernelGGL(stats_finalize, dim3((N + 255) / 256), dim3(256),
                           0, stream, psum, psq, g, beta, B, N, EPS_BN, scale, shift);
    };
    auto bnrelu = [&](const unsigned short* Y, unsigned short* O, int N) {
        long total = (long)B * N;
        hipLaunchKernelGGL(bn_relu_bf16, dim3((total + 255) / 256), dim3(256),
                           0, stream, Y, N, scale, shift, O, total);
    };

    // ---- image branch (p) ----
    gemm_f32a(img_p, 300, 300, wWi1, 320, bi1, act0, 0, 1024, 1024, 320);
    bnstats(act0, 0, 1024, 1024, gi1, bti1);
    bnrelu(act0, act1, 1024);
    gemm_async(act1, 1024, wWi2, 1024, bi2, act0, 0, 1024, 1024, 1024);
    gemm_async(act0, 1024, wWi3, 1024, bi3, outP, 1, 300, 300, 1024);

    // ---- image branch (n): separate batch stats, same weights ----
    gemm_f32a(img_n, 300, 300, wWi1, 320, bi1, act0, 0, 1024, 1024, 320);
    bnstats(act0, 0, 1024, 1024, gi1, bti1);
    bnrelu(act0, act1, 1024);
    gemm_async(act1, 1024, wWi2, 1024, bi2, act0, 0, 1024, 1024, 1024);
    gemm_async(act0, 1024, wWi3, 1024, bi3, outN, 1, 300, 300, 1024);

    // ---- multimodal branch ----
    hipLaunchKernelGGL(loc_fc, dim3((B + 255) / 256), dim3(256), 0, stream,
                       lat, lon, Wloc, bloc, locy, B);
    bnstats(locy, 1, 10, 10, gloc, btloc);
    {
        long total = (long)B * 320;
        hipLaunchKernelGGL(build_a, dim3((total + 255) / 256), dim3(256), 0, stream,
                           locy, scale, shift, tag, abuf, B);
    }
    gemm_async(abuf, 320, wWm1, 320, bm1, act0, 0, 1024, 1024, 320);
    bnstats(act0, 0, 1024, 1024, gm1, btm1);
    bnrelu(act0, act1, 1024);
    gemm_async(act1, 1024, wWm2, 1024, bm2, act0, 0, 1024, 1024, 1024);
    bnstats(act0, 0, 1024, 1024, gm2, btm2);
    bnrelu(act0, act1, 1024);
    gemm_async(act1, 1024, wWm3, 1024, bm3, act0, 0, 1024, 1024, 1024);
    bnstats(act0, 0, 1024, 1024, gm3, btm3);
    bnrelu(act0, act1, 1024);
    gemm_async(act1, 1024, wWm4, 1024, bm4, outA, 1, 300, 300, 1024);

    // ---- triplet margin count ----
    hipLaunchKernelGGL(init_count, dim3(1), dim3(64), 0, stream, cnt);
    hipLaunchKernelGGL(tri_count, dim3(B / 8), dim3(256), 0, stream,
                       (const float*)d_out, B, cnt);
}
